// VectorQuantizer_21337397527271
// MI455X (gfx1250) — matrix operand order repaired and checked by reading, not compiled, not run on hardware
//
#include <hip/hip_runtime.h>
#include <stdint.h>

typedef __bf16    v16bf __attribute__((ext_vector_type(16)));
typedef float     v8f   __attribute__((ext_vector_type(8)));
typedef uint32_t  v8u   __attribute__((ext_vector_type(8)));

#define KCODES 1024
#define DIM    64
#define NVEC   131072      // 32*64*64 rows
#define HWSZ   4096        // 64*64
#define NUMEL  8388608     // 32*64*64*64
#define BETA_F 0.25f
#define RT     4           // row tiles (16 rows each) per wave

// ---- workspace layout (bytes) ----
#define WS_IDX       0                         // int[NVEC]
#define WS_USAGE     (WS_IDX + NVEC * 4)       // int[KCODES]
#define WS_SUMSQ     (WS_USAGE + KCODES * 4)   // float[1]
#define WS_CODENORM  (WS_SUMSQ + 256)          // float[KCODES]
#define WS_B         (WS_CODENORM + KCODES * 4)// uint32[64*2*32*8] packed bf16 B tiles

static __device__ __forceinline__ uint32_t pack_bf16(float a, float b) {
    uint32_t ua = __builtin_bit_cast(uint32_t, a);
    uint32_t ub = __builtin_bit_cast(uint32_t, b);
    ua = (ua + 0x7FFFu + ((ua >> 16) & 1u)) >> 16;   // RNE f32 -> bf16
    ub = (ub + 0x7FFFu + ((ub >> 16) & 1u)) >> 16;
    return (ua & 0xFFFFu) | (ub << 16);
}

// order-preserving f32 -> u32 key, low 10 bits replaced by candidate index
static __device__ __forceinline__ uint32_t dist_key(float d, uint32_t cand) {
    uint32_t b    = __builtin_bit_cast(uint32_t, d);
    uint32_t mask = (uint32_t)(((int32_t)b) >> 31) | 0x80000000u;
    return ((b ^ mask) & 0xFFFFFC00u) | cand;
}

// ---------------------------------------------------------------------------
// Kernel 0: zero accumulators, codebook norms, pack W into WMMA B-layout bf16
// tid = t*64 + kk*32 + lane : tile t (0..63), k-half kk (0..1), lane (0..31)
// B layout (16-bit B 32x16): lane n = lane&15, K = kk*32 + (lane>>4)*16 + 2v,2v+1
// ---------------------------------------------------------------------------
__global__ __launch_bounds__(256) void vq_prep(const float* __restrict__ W,
                                               uint32_t* __restrict__ wsB,
                                               float* __restrict__ codenorm,
                                               int* __restrict__ usage,
                                               float* __restrict__ sumsq) {
    int tid  = blockIdx.x * 256 + threadIdx.x;          // 0..4095
    int t    = tid >> 6;
    int kk   = (tid >> 5) & 1;
    int lane = tid & 31;
    int n    = lane & 15;
    int half = lane >> 4;
    int code = t * 16 + n;
    int koff = kk * 32 + half * 16;
    const float* wr = W + code * DIM + koff;
    uint32_t* dst = wsB + (size_t)tid * 8;
#pragma unroll
    for (int v = 0; v < 8; ++v)
        dst[v] = pack_bf16(wr[2 * v], wr[2 * v + 1]);

    if (tid < KCODES) {
        const float* p = W + tid * DIM;
        float s = 0.f;
#pragma unroll
        for (int d = 0; d < DIM; ++d) s += p[d] * p[d];
        codenorm[tid] = s;
        usage[tid]    = 0;
    }
    if (tid == 0) *sumsq = 0.f;
}

// ---------------------------------------------------------------------------
// Kernel 1: WMMA distance + argmin. One wave handles RT row-tiles (64 rows),
// sharing each codebook B-tile load across RT WMMAs (4x less L2 traffic).
// A (16-bit 16x32): lane m = lane&15; VGPR v holds K=(lane>>4)*16+2v
// C (f32 16x16):    lane l -> N = l&15, VGPR j -> M = j + 8*(l>>4)
// ---------------------------------------------------------------------------
__global__ __launch_bounds__(256) void vq_argmin(const float* __restrict__ ze,
                                                 const uint32_t* __restrict__ wsB,
                                                 const float* __restrict__ codenorm,
                                                 int* __restrict__ ws_idx,
                                                 int* __restrict__ usage,
                                                 float* __restrict__ out_idx) {
    int lane     = threadIdx.x & 31;
    int wave     = threadIdx.x >> 5;
    int rowbase0 = (blockIdx.x * 8 + wave) * (16 * RT);
    int m        = lane & 15;
    int half     = lane >> 4;

    // load + pack A operands for the RT row tiles
    v16bf a[RT][2];
#pragma unroll
    for (int rt = 0; rt < RT; ++rt) {
        int r  = rowbase0 + rt * 16 + m;
        int b  = r >> 12;          // batch
        int hw = r & 4095;         // spatial position
        const float* zrow = ze + (size_t)b * (DIM * HWSZ) + hw;  // dim d at +d*HWSZ
        v8u u0, u1;
#pragma unroll
        for (int v = 0; v < 8; ++v) {
            int kpair = half * 16 + 2 * v;
            u0[v] = pack_bf16(zrow[(size_t)kpair * HWSZ],
                              zrow[(size_t)(kpair + 1) * HWSZ]);
            u1[v] = pack_bf16(zrow[(size_t)(kpair + 32) * HWSZ],
                              zrow[(size_t)(kpair + 33) * HWSZ]);
        }
        a[rt][0] = __builtin_bit_cast(v16bf, u0);
        a[rt][1] = __builtin_bit_cast(v16bf, u1);
    }

    uint32_t bkey[RT][8];
#pragma unroll
    for (int rt = 0; rt < RT; ++rt)
#pragma unroll
        for (int j = 0; j < 8; ++j) bkey[rt][j] = 0xFFFFFFFFu;

    for (int t = 0; t < 64; ++t) {
        v8u b0u = *(const v8u*)(wsB + (size_t)(t * 64 + lane) * 8);
        v8u b1u = *(const v8u*)(wsB + (size_t)(t * 64 + 32 + lane) * 8);
        v16bf b0 = __builtin_bit_cast(v16bf, b0u);
        v16bf b1 = __builtin_bit_cast(v16bf, b1u);
        float    cn   = codenorm[t * 16 + m];
        uint32_t cand = (uint32_t)(t * 16 + m);

#pragma unroll
        for (int rt = 0; rt < RT; ++rt) {
            v8f c = {0.f, 0.f, 0.f, 0.f, 0.f, 0.f, 0.f, 0.f};
            c = __builtin_amdgcn_wmma_f32_16x16x32_bf16(false, a[rt][0], false, b0,
                                                        (short)0, c, false, false);
            c = __builtin_amdgcn_wmma_f32_16x16x32_bf16(false, a[rt][1], false, b1,
                                                        (short)0, c, false, false);
#pragma unroll
            for (int j = 0; j < 8; ++j) {
                float dj = fmaf(-2.0f, c[j], cn);
                uint32_t key = dist_key(dj, cand);
                bkey[rt][j] = (key < bkey[rt][j]) ? key : bkey[rt][j];
            }
        }
    }

    // min-reduce keys across the 16 N-lanes of each half (wave32 butterfly)
#pragma unroll
    for (int off = 1; off < 16; off <<= 1) {
#pragma unroll
        for (int rt = 0; rt < RT; ++rt)
#pragma unroll
            for (int j = 0; j < 8; ++j) {
                uint32_t ok = __shfl_xor(bkey[rt][j], off, 32);
                bkey[rt][j] = (ok < bkey[rt][j]) ? ok : bkey[rt][j];
            }
    }

    if (m == 0) {   // lanes 0 and 16 hold rows (rt*16 + half*8 + j)
#pragma unroll
        for (int rt = 0; rt < RT; ++rt)
#pragma unroll
            for (int j = 0; j < 8; ++j) {
                int row = rowbase0 + rt * 16 + half * 8 + j;
                int k   = (int)(bkey[rt][j] & 1023u);
                ws_idx[row]  = k;
                out_idx[row] = (float)k;
                atomicAdd(&usage[k], 1);
            }
    }
}

// ---------------------------------------------------------------------------
// Kernel 2: gather zq (== zq_st numerically) in [B,D,H,W] layout + MSE accum
// ---------------------------------------------------------------------------
__global__ __launch_bounds__(256) void vq_gather(const float* __restrict__ ze,
                                                 const float* __restrict__ W,
                                                 const int* __restrict__ ws_idx,
                                                 float* __restrict__ out_zq,
                                                 float* __restrict__ sumsq) {
    size_t base = ((size_t)blockIdx.x * 256 + threadIdx.x) * 4;
    int hw = (int)(base & 4095);
    int d  = (int)((base >> 12) & 63);
    int b  = (int)(base >> 18);
    int n  = b * HWSZ + hw;

    float4 zv = *(const float4*)(ze + base);
    int k0 = ws_idx[n + 0], k1 = ws_idx[n + 1];
    int k2 = ws_idx[n + 2], k3 = ws_idx[n + 3];
    float q0 = W[k0 * DIM + d], q1 = W[k1 * DIM + d];
    float q2 = W[k2 * DIM + d], q3 = W[k3 * DIM + d];
    float4 qv = make_float4(q0, q1, q2, q3);
    *(float4*)(out_zq + base) = qv;

    float d0 = q0 - zv.x, d1 = q1 - zv.y, d2 = q2 - zv.z, d3 = q3 - zv.w;
    float local = d0 * d0 + d1 * d1 + d2 * d2 + d3 * d3;

    __shared__ float red[256];
    red[threadIdx.x] = local;
    __syncthreads();
#pragma unroll
    for (int s = 128; s > 0; s >>= 1) {
        if ((int)threadIdx.x < s) red[threadIdx.x] += red[threadIdx.x + s];
        __syncthreads();
    }
    if (threadIdx.x == 0) atomicAdd(sumsq, red[0]);
}

// ---------------------------------------------------------------------------
// Kernel 3: usage -> float, perplexity, dead count, scalar losses
// ---------------------------------------------------------------------------
__global__ __launch_bounds__(1024) void vq_finalize(const int* __restrict__ usage,
                                                    const float* __restrict__ sumsq,
                                                    float* __restrict__ out_scalars,
                                                    float* __restrict__ out_usage) {
    __shared__ float sh[1024];
    __shared__ int   shd[1024];
    int k = threadIdx.x;
    int u = usage[k];
    out_usage[k] = (float)u;
    float p = (float)u / (float)NVEC;
    sh[k]  = (p > 0.f) ? p * logf(p) : 0.f;
    shd[k] = (u < 2) ? 1 : 0;
    __syncthreads();
#pragma unroll
    for (int s = 512; s > 0; s >>= 1) {
        if (k < s) { sh[k] += sh[k + s]; shd[k] += shd[k + s]; }
        __syncthreads();
    }
    if (k == 0) {
        float mse = *sumsq / (float)NUMEL;
        out_scalars[0] = mse + BETA_F * mse;  // q_loss (codebook + beta*commitment)
        out_scalars[1] = mse;                 // codebook_loss
        out_scalars[2] = mse;                 // commitment_loss
        out_scalars[3] = expf(-sh[0]);        // perplexity
        out_scalars[4] = (float)shd[0];       // dead codes
    }
}

// ---------------------------------------------------------------------------
extern "C" void kernel_launch(void* const* d_in, const int* in_sizes, int n_in,
                              void* d_out, int out_size, void* d_ws, size_t ws_size,
                              hipStream_t stream) {
    const float* ze = (const float*)d_in[0];   // [32,64,64,64] f32
    const float* W  = (const float*)d_in[1];   // [1024,64] f32
    float* out = (float*)d_out;

    char* ws = (char*)d_ws;
    int*      ws_idx   = (int*)(ws + WS_IDX);
    int*      usage    = (int*)(ws + WS_USAGE);
    float*    sumsq    = (float*)(ws + WS_SUMSQ);
    float*    codenorm = (float*)(ws + WS_CODENORM);
    uint32_t* wsB      = (uint32_t*)(ws + WS_B);

    float* out_zq      = out;                       // NUMEL
    float* out_idx     = out + NUMEL;               // NVEC
    float* out_scalars = out + NUMEL + NVEC;        // 5 scalars
    float* out_usage   = out_scalars + 5;           // KCODES

    vq_prep    <<<16,  256, 0, stream>>>(W, wsB, codenorm, usage, sumsq);
    vq_argmin  <<<256, 256, 0, stream>>>(ze, wsB, codenorm, ws_idx, usage, out_idx);
    vq_gather  <<<8192,256, 0, stream>>>(ze, W, ws_idx, out_zq, sumsq);
    vq_finalize<<<1,  1024, 0, stream>>>(usage, sumsq, out_scalars, out_usage);
}
